// SleepStageGNN_32461362823950
// MI455X (gfx1250) — matrix operand order repaired and checked by reading, not compiled, not run on hardware
//
#include <hip/hip_runtime.h>
#include <math.h>

typedef __attribute__((ext_vector_type(16))) _Float16 v16h;
typedef __attribute__((ext_vector_type(8)))  _Float16 v8h;
typedef __attribute__((ext_vector_type(8)))  float    v8f;

#define DEVI __device__ __forceinline__

DEVI float sigf(float x){ return 1.0f / (1.0f + __expf(-x)); }

DEVI v16h cat8(v8h lo, v8h hi){
  return __builtin_shufflevector(lo, hi, 0,1,2,3,4,5,6,7,8,9,10,11,12,13,14,15);
}

// ---------------- WMMA fragment loads (wave32, v_wmma_f32_16x16x32_f16) -------
// A (16x32 f16) from row-major [m][k] panel, strideH halves (mult of 8):
//   lane L: row m0+(L&15); halves 0..7 = k0+khi+0..7, halves 8..15 = k0+16+khi+0..7
//   -> two contiguous 16B loads (ds_load_b128 / global_load_b128)
DEVI v16h frag_a_h(const _Float16* A, int strideH, int m0, int k0, int lane){
  const _Float16* p = A + (size_t)(m0 + (lane & 15)) * strideH
                        + k0 + ((lane >> 4) << 3);
  return cat8(*(const v8h*)p, *(const v8h*)(p + 16));
}
// B (32x16 f16) from row-major [n][k] matrix: lane L: col n0+(L&15),
//   halves 0..7 = k0+khi+0..7, halves 8..15 = k0+16+khi+0..7 -> two 16B loads
DEVI v16h frag_b_h(const _Float16* B, int strideH, int k0, int n0, int lane){
  const _Float16* p = B + (size_t)(n0 + (lane & 15)) * strideH
                        + k0 + ((lane >> 4) << 3);
  return cat8(*(const v8h*)p, *(const v8h*)(p + 16));
}
// C/D (16x16 f32, v8f): VGPR j -> row m0 + j + 8*(lane>=16), col n0 + (lane&15)

// =====================================================================
// Kernel 0: f32 -> f16 weight conversion (row-major preserved)
// =====================================================================
__global__ __launch_bounds__(256)
void cvt_f16_kernel(const float* __restrict__ in, _Float16* __restrict__ out, int n){
  int i = blockIdx.x*256 + threadIdx.x;
  if (i < n) out[i] = (_Float16)in[i];
}

// =====================================================================
// Kernel 1: adjacency normalization (tiny)
// =====================================================================
__global__ void build_adj_kernel(const int* __restrict__ ei, float* __restrict__ adj){
  if (threadIdx.x == 0 && blockIdx.x == 0){
    float A[8][8];
    for (int i = 0; i < 8; ++i)
      for (int j = 0; j < 8; ++j) A[i][j] = (i == j) ? 1.0f : 0.0f;
    for (int e = 0; e < 56; ++e) A[ei[56 + e]][ei[e]] += 1.0f;
    float dinv[8];
    for (int i = 0; i < 8; ++i){
      float sum = 0.0f;
      for (int j = 0; j < 8; ++j) sum += A[i][j];
      dinv[i] = rsqrtf(sum);
    }
    for (int i = 0; i < 8; ++i)
      for (int j = 0; j < 8; ++j) adj[i*8 + j] = A[i][j] * dinv[i] * dinv[j];
  }
}

// =====================================================================
// Kernel 2: fused conv stack, one block = one sequence (10240 blocks).
// LDS time-multiplexed regions:
//   R1 [0,49152):  xs f32[3008] -> A2 f16[256][96] -> c2 f32[32][256] -> A3 f16[96][96]
//   R2 [49152,+16384): h1 f16[16][512] -> h2 f16[32][96]
//   R3 [65536,+6144):  w2h f16[32][96]   R3b [71680,+12288): w3h f16[64][96]
//   R4 [83968,+256):   accf f32[64]
// =====================================================================
#define CONV_SMEM (49152 + 16384 + 6144 + 12288 + 256)

__global__ __launch_bounds__(256)
void conv_stack_kernel(const float* __restrict__ x,
    const float* __restrict__ w1, const float* __restrict__ b1c,
    const float* __restrict__ g1bn, const float* __restrict__ b1bn,
    const float* __restrict__ w2, const float* __restrict__ b2c,
    const float* __restrict__ g2bn, const float* __restrict__ b2bn,
    const float* __restrict__ w3, const float* __restrict__ b3c,
    const float* __restrict__ g3bn, const float* __restrict__ b3bn,
    float* __restrict__ feats)
{
  extern __shared__ char smem[];
  float*    xs  = (float*)smem;                    // padded input: [0..3]=0, data at +4
  _Float16* A2  = (_Float16*)smem;                 // [256][96]
  float*    c2  = (float*)smem;                    // [32][256]
  _Float16* A3  = (_Float16*)smem;                 // [96][96]
  _Float16* h1  = (_Float16*)(smem + 49152);       // [16][512]
  _Float16* h2  = (_Float16*)(smem + 49152);       // [32][96] (h1 dead by then)
  _Float16* w2h = (_Float16*)(smem + 65536);       // [32][96] zero-padded K
  _Float16* w3h = (_Float16*)(smem + 71680);       // [64][96]
  float*   accf = (float*)(smem + 83968);

  const int tid = threadIdx.x, lane = tid & 31, wv = tid >> 5;
  const int s = blockIdx.x;
  const float bns = rsqrtf(1.0f + 1e-5f);          // bn_eval: / sqrt(1+EPS)

  // ---- stage input (4-float guard bands: branch-free conv1) + weights ----
  const float* xrow = x + (size_t)s * 3000;
  for (int i = tid; i < 3008; i += 256){
    float v = 0.0f;
    if (i >= 4 && i < 3004) v = xrow[i - 4];
    if ((i & 63) == 0) __builtin_prefetch(xrow + i + 512, 0, 1);
    xs[i] = v;
  }
  for (int idx = tid; idx < 32*96; idx += 256){
    int n = idx / 96, k = idx - n*96;
    w2h[idx] = (k < 80) ? (_Float16)w2[n*80 + k] : (_Float16)0.0f;
  }
  for (int i = tid; i < 6144; i += 256) w3h[i] = (_Float16)w3[i];   // [64][96] exact
  if (tid < 64) accf[tid] = 0.0f;
  __syncthreads();

  // ---- conv1(k7,s2,p3)+bn+relu+pool3 -> h1 f16 [16][512] (500 real, pads 0) ----
  for (int idx = tid; idx < 16*500; idx += 256){
    int c = idx / 500, p = idx - c*500;
    float sc = g1bn[c]*bns, sh = b1bn[c], cb = b1c[c];
    float m = -1e30f;
#pragma unroll
    for (int sub = 0; sub < 3; ++sub){
      const float* xp = xs + 2*(3*p + sub) + 1;    // +4 guard -3 pad
      float a = cb;
#pragma unroll
      for (int k = 0; k < 7; ++k) a += xp[k] * w1[c*7 + k];
      float v = a*sc + sh; v = v > 0.0f ? v : 0.0f;
      m = v > m ? v : m;
    }
    h1[c*512 + p] = (_Float16)m;
  }
  if (tid < 16*12){ int c = tid/12, j = tid - c*12; h1[c*512 + 500 + j] = (_Float16)0.0f; }
  __syncthreads();

  // ---- stage im2col panel A2[256][96] (k = ci*5+kk, zero-padded 80..95) ----
  // h1 pad columns are zero, so (t & 511) absorbs the conv boundary.
  for (int idx = tid; idx < 256*96; idx += 256){
    int m = idx / 96, k = idx - m*96;
    int kc = (k < 80) ? k : 0;
    int ci = kc / 5, r = kc - ci*5;
    int t = 2*m + r - 2;
    _Float16 v = h1[ci*512 + (t & 511)];
    A2[idx] = (k < 80) ? v : (_Float16)0.0f;
  }
  __syncthreads();

  // ---- conv2 WMMA GEMM: M=256 pos, N=32 ch, K=96 -> regs ----
  v8f acc2[4];
#pragma unroll
  for (int i = 0; i < 4; ++i){
    int q = wv*4 + i, m0 = (q >> 1)*16, n0 = (q & 1)*16;
    v8f acc = {};
#pragma unroll
    for (int kk = 0; kk < 3; ++kk){
      v16h af = frag_a_h(A2,  96, m0, kk*32, lane);
      v16h bf = frag_b_h(w2h, 96, kk*32, n0, lane);
      acc = __builtin_amdgcn_wmma_f32_16x16x32_f16(false, af, false, bf,
                                                   (short)0, acc, false, false);
    }
    acc2[i] = acc;
  }
  __syncthreads();     // all waves done reading A2 -> reuse region as c2
  // ---- epilogue: bias+bn+relu -> c2 f32 [32][256] ----
#pragma unroll
  for (int i = 0; i < 4; ++i){
    int q = wv*4 + i, m0 = (q >> 1)*16, n0 = (q & 1)*16;
    int n = n0 + (lane & 15);
    float sc = g2bn[n]*bns, sh = b2bn[n], cb = b2c[n];
#pragma unroll
    for (int j = 0; j < 8; ++j){
      int m = m0 + j + ((lane >> 4) << 3);
      float v = (acc2[i][j] + cb)*sc + sh;
      c2[n*256 + m] = v > 0.0f ? v : 0.0f;
    }
  }
  __syncthreads();

  // ---- pool3 -> h2 f16 [32][96] (83 real, rest 0) ----
  for (int idx = tid; idx < 32*96; idx += 256){
    int c = idx / 96, p = idx - c*96;
    float v = 0.0f;
    if (p < 83){
      const float* r = c2 + c*256 + 3*p;
      v = fmaxf(r[0], fmaxf(r[1], r[2]));
    }
    h2[c*96 + p] = (_Float16)v;
  }
  __syncthreads();

  // ---- stage im2col panel A3[96][96] (k = ci*3+kk) ----
  for (int idx = tid; idx < 96*96; idx += 256){
    int m = idx / 96, k = idx - m*96;
    int ci = k / 3, r = k - ci*3;
    int t = m + r - 1;
    int tc = t < 0 ? 0 : (t > 95 ? 95 : t);
    _Float16 v = h2[ci*96 + tc];
    A3[idx] = (t == tc) ? v : (_Float16)0.0f;
  }
  __syncthreads();

  // ---- conv3 WMMA GEMM: M=96 pos (83 real), N=64 ch, K=96; fused mean ----
#pragma unroll
  for (int i = 0; i < 3; ++i){
    int q = wv*3 + i, m0 = (q >> 2)*16, n0 = (q & 3)*16;
    v8f acc = {};
#pragma unroll
    for (int kk = 0; kk < 3; ++kk){
      v16h af = frag_a_h(A3,  96, m0, kk*32, lane);
      v16h bf = frag_b_h(w3h, 96, kk*32, n0, lane);
      acc = __builtin_amdgcn_wmma_f32_16x16x32_f16(false, af, false, bf,
                                                   (short)0, acc, false, false);
    }
    int n = n0 + (lane & 15);
    float sc = g3bn[n]*bns, sh = b3bn[n], cb = b3c[n];
    float part = 0.0f;
#pragma unroll
    for (int j = 0; j < 8; ++j){
      int m = m0 + j + ((lane >> 4) << 3);
      if (m < 83){
        float v = (acc[j] + cb)*sc + sh;
        part += v > 0.0f ? v : 0.0f;
      }
    }
    atomicAdd(&accf[n], part);
  }
  __syncthreads();
  if (tid < 64) feats[(size_t)s*64 + tid] = accf[tid] * (1.0f/83.0f);
}

// =====================================================================
// Kernel 3: GCN (8-node graph). One block = one (b,s) sample; VALU f32
// (matrices are 8xF -- too small for 16x16 tiles; <2% of total FLOPs).
// Emits emb directly in f16 [1280][128] for the WMMA input projection.
// =====================================================================
__global__ __launch_bounds__(128)
void gcn_kernel(const float* __restrict__ feats, const float* __restrict__ adj,
    const float* __restrict__ w0, const float* __restrict__ b0,
    const float* __restrict__ w1, const float* __restrict__ b1,
    const float* __restrict__ lng, const float* __restrict__ lnb,
    _Float16* __restrict__ emb_h)
{
  __shared__ float F[8][64];
  __shared__ float An[64];
  __shared__ float g1s[8][128];
  __shared__ float g2s[8][128];
  __shared__ float mv[16];
  const int tid = threadIdx.x, s = blockIdx.x;

  for (int i = tid; i < 512; i += 128) ((float*)F)[i] = feats[(size_t)s*512 + i];
  if (tid < 64) An[tid] = adj[tid];
  __syncthreads();

  const int n = tid;
  float t0[8];
#pragma unroll
  for (int r = 0; r < 8; ++r){
    float a = 0.0f;
    for (int j = 0; j < 64; ++j) a += F[r][j] * w0[n*64 + j];
    t0[r] = a;
  }
#pragma unroll
  for (int r = 0; r < 8; ++r){
    float a = b0[n];
    for (int j = 0; j < 8; ++j) a += An[r*8 + j] * t0[j];
    g1s[r][n] = a > 0.0f ? a : 0.0f;
  }
  __syncthreads();
  float t1[8];
#pragma unroll
  for (int r = 0; r < 8; ++r){
    float a = 0.0f;
    for (int j = 0; j < 128; ++j) a += g1s[r][j] * w1[n*128 + j];
    t1[r] = a;
  }
#pragma unroll
  for (int r = 0; r < 8; ++r){
    float a = b1[n];
    for (int j = 0; j < 8; ++j) a += An[r*8 + j] * t1[j];
    g2s[r][n] = a;
  }
  __syncthreads();
  if (tid < 8){
    float m = 0.0f;
    for (int j = 0; j < 128; ++j) m += g2s[tid][j];
    m *= (1.0f/128.0f);
    float v = 0.0f;
    for (int j = 0; j < 128; ++j){ float d = g2s[tid][j] - m; v += d*d; }
    v *= (1.0f/128.0f);
    mv[tid] = m; mv[8 + tid] = rsqrtf(v + 1e-5f);
  }
  __syncthreads();
  float e = 0.0f;
#pragma unroll
  for (int r = 0; r < 8; ++r) e += (g2s[r][n] - mv[r]) * mv[8 + r];
  emb_h[(size_t)s*128 + n] = (_Float16)((e * (1.0f/8.0f)) * lng[n] + lnb[n]);
}

// =====================================================================
// Kernel 4: LSTM input projection  xs = A(1280,K) @ W(1024,K)^T + (b_ih+b_hh)
// A and W are pre-converted f16 row-major -> fragments are pure b128 loads.
// 8 waves/block (2x4 tiles); grid (40,16).
// =====================================================================
__global__ __launch_bounds__(256)
void xproj_kernel(const _Float16* __restrict__ A, const _Float16* __restrict__ W,
                  const float* __restrict__ bih, const float* __restrict__ bhh,
                  float* __restrict__ out, int K)
{
  const int tid = threadIdx.x, lane = tid & 31, wv = tid >> 5;
  const int m0 = blockIdx.x*32 + (wv >> 2)*16;
  const int n0 = blockIdx.y*64 + (wv & 3)*16;
  v8f acc = {};
  for (int k0 = 0; k0 < K; k0 += 32){
    v16h af = frag_a_h(A, K, m0, k0, lane);
    v16h bf = frag_b_h(W, K, k0, n0, lane);
    acc = __builtin_amdgcn_wmma_f32_16x16x32_f16(false, af, false, bf,
                                                 (short)0, acc, false, false);
  }
  const int n = n0 + (lane & 15);
  const float bb = bih[n] + bhh[n];
#pragma unroll
  for (int j = 0; j < 8; ++j){
    int m = m0 + j + ((lane >> 4) << 3);
    out[(size_t)m*1024 + n] = acc[j] + bb;
  }
}

// =====================================================================
// Kernel 5: LSTM recurrence. 2 blocks (fwd/rev), 1024 threads (32 waves).
// h f16 + c f32 persistent in LDS; per-step h@W_hh^T via WMMA (W_hh pre-f16,
// L2-resident); gates staged through ws between GEMM and pointwise phases.
// Dual output: f32 (for LN head) + f16 (for next layer's projection).
// =====================================================================
#define LSTM_SMEM (32768 + 65536)

__global__ __launch_bounds__(1024)
void lstm_rec_kernel(const float* __restrict__ xs_f, const float* __restrict__ xs_r,
                     const _Float16* __restrict__ whh_fh, const _Float16* __restrict__ whh_rh,
                     float* __restrict__ gbuf_base,
                     float* __restrict__ out, _Float16* __restrict__ outh)
{
  extern __shared__ char smem[];
  _Float16* hS = (_Float16*)smem;             // [64][256]
  float*    cS = (float*)(smem + 32768);      // [64][256]
  const int dir = blockIdx.x;
  const float*    xs  = dir ? xs_r  : xs_f;
  const _Float16* whh = dir ? whh_rh : whh_fh;
  float* gbuf = gbuf_base + (size_t)dir * 64 * 1024;
  const int tid = threadIdx.x, lane = tid & 31, wv = tid >> 5;

  for (int i = tid; i < 64*256; i += 1024){ hS[i] = (_Float16)0.0f; cS[i] = 0.0f; }
  __syncthreads();

  for (int st = 0; st < 20; ++st){
    const int tt = dir ? (19 - st) : st;
    // --- phase 1: gates_h = h @ W_hh^T (WMMA, 8 tiles/wave) -> gbuf ---
    for (int i = 0; i < 8; ++i){
      int q = wv*8 + i, m0 = (q >> 6)*16, n0 = (q & 63)*16;
      v8f acc = {};
#pragma unroll
      for (int kk = 0; kk < 8; ++kk){
        v16h af = frag_a_h(hS,  256, m0, kk*32, lane);   // ds_load_b128 x2
        v16h bf = frag_b_h(whh, 256, kk*32, n0, lane);   // global_load_b128 x2
        acc = __builtin_amdgcn_wmma_f32_16x16x32_f16(false, af, false, bf,
                                                     (short)0, acc, false, false);
      }
      int n = n0 + (lane & 15);
#pragma unroll
      for (int j = 0; j < 8; ++j){
        int m = m0 + j + ((lane >> 4) << 3);
        gbuf[m*1024 + n] = acc[j];
      }
    }
    __threadfence_block();
    __syncthreads();
    // --- phase 2: pointwise gates, update c/h, emit outputs ---
    for (int cell = tid; cell < 64*256; cell += 1024){
      int b = cell >> 8, u = cell & 255;
      size_t xrow = ((size_t)b*20 + tt) * 1024;
      float gi = xs[xrow +       u] + gbuf[b*1024 +       u];
      float gf = xs[xrow + 256 + u] + gbuf[b*1024 + 256 + u];
      float gg = xs[xrow + 512 + u] + gbuf[b*1024 + 512 + u];
      float go = xs[xrow + 768 + u] + gbuf[b*1024 + 768 + u];
      float c = sigf(gf)*cS[cell] + sigf(gi)*tanhf(gg);
      float h = sigf(go)*tanhf(c);
      cS[cell] = c;
      hS[cell] = (_Float16)h;
      size_t orow = ((size_t)b*20 + tt)*512 + dir*256 + u;
      out[orow]  = h;
      outh[orow] = (_Float16)h;
    }
    __threadfence_block();
    __syncthreads();
  }
}

// =====================================================================
// Kernel 6: final LayerNorm + classifier. One block = one (b,t) row.
// =====================================================================
__global__ __launch_bounds__(128)
void head_kernel(const float* __restrict__ l1, const float* __restrict__ lng,
                 const float* __restrict__ lnb, const float* __restrict__ cw,
                 const float* __restrict__ cb, float* __restrict__ out)
{
  __shared__ float xn[512];
  __shared__ float red[128];
  const int tid = threadIdx.x, row = blockIdx.x;
  const float* xr = l1 + (size_t)row*512;
  float lsum = 0.0f, lsq = 0.0f;
  for (int i = tid; i < 512; i += 128){ float v = xr[i]; lsum += v; lsq += v*v; }
  red[tid] = lsum; __syncthreads();
  for (int o = 64; o > 0; o >>= 1){ if (tid < o) red[tid] += red[tid + o]; __syncthreads(); }
  float mean = red[0] * (1.0f/512.0f);
  __syncthreads();
  red[tid] = lsq; __syncthreads();
  for (int o = 64; o > 0; o >>= 1){ if (tid < o) red[tid] += red[tid + o]; __syncthreads(); }
  float var = red[0] * (1.0f/512.0f) - mean*mean;
  float inv = rsqrtf(var + 1e-5f);
  for (int i = tid; i < 512; i += 128) xn[i] = (xr[i] - mean)*inv*lng[i] + lnb[i];
  __syncthreads();
  if (tid < 5){
    float a = cb[tid];
    for (int j = 0; j < 512; ++j) a += xn[j] * cw[tid*512 + j];
    out[(size_t)row*5 + tid] = a;
  }
}

// =====================================================================
extern "C" void kernel_launch(void* const* d_in, const int* in_sizes, int n_in,
                              void* d_out, int out_size, void* d_ws, size_t ws_size,
                              hipStream_t stream) {
  (void)in_sizes; (void)n_in; (void)out_size; (void)ws_size;
  const float* x        = (const float*)d_in[0];
  const int*   edge     = (const int*)  d_in[1];
  const float* conv1_w  = (const float*)d_in[2];
  const float* conv1_b  = (const float*)d_in[3];
  const float* bn1_g    = (const float*)d_in[4];
  const float* bn1_b    = (const float*)d_in[5];
  const float* conv2_w  = (const float*)d_in[6];
  const float* conv2_b  = (const float*)d_in[7];
  const float* bn2_g    = (const float*)d_in[8];
  const float* bn2_b    = (const float*)d_in[9];
  const float* conv3_w  = (const float*)d_in[10];
  const float* conv3_b  = (const float*)d_in[11];
  const float* bn3_g    = (const float*)d_in[12];
  const float* bn3_b    = (const float*)d_in[13];
  const float* gcn0_w   = (const float*)d_in[14];
  const float* gcn0_b   = (const float*)d_in[15];
  const float* gcn1_w   = (const float*)d_in[16];
  const float* gcn1_b   = (const float*)d_in[17];
  const float* gln_g    = (const float*)d_in[18];
  const float* gln_b    = (const float*)d_in[19];
  const float* w_ih_l0  = (const float*)d_in[20];
  const float* w_hh_l0  = (const float*)d_in[21];
  const float* b_ih_l0  = (const float*)d_in[22];
  const float* b_hh_l0  = (const float*)d_in[23];
  const float* w_ih_l0r = (const float*)d_in[24];
  const float* w_hh_l0r = (const float*)d_in[25];
  const float* b_ih_l0r = (const float*)d_in[26];
  const float* b_hh_l0r = (const float*)d_in[27];
  const float* w_ih_l1  = (const float*)d_in[28];
  const float* w_hh_l1  = (const float*)d_in[29];
  const float* b_ih_l1  = (const float*)d_in[30];
  const float* b_hh_l1  = (const float*)d_in[31];
  const float* w_ih_l1r = (const float*)d_in[32];
  const float* w_hh_l1r = (const float*)d_in[33];
  const float* b_ih_l1r = (const float*)d_in[34];
  const float* b_hh_l1r = (const float*)d_in[35];
  const float* lln_g    = (const float*)d_in[36];
  const float* lln_b    = (const float*)d_in[37];
  const float* cls_w    = (const float*)d_in[38];
  const float* cls_b    = (const float*)d_in[39];

  // -------- workspace carve (256B-aligned; ~27 MB, L2-resident) --------
  char* wsp = (char*)d_ws;
  size_t off = 0;
#define CARVE(TY, NAME, BYTES) TY NAME = (TY)(wsp + off); off += (((size_t)(BYTES)) + 255) & ~(size_t)255;
  CARVE(float*,     adj,    256)
  CARVE(float*,     feats,  10240*64*4)
  CARVE(_Float16*,  emb_h,  1280*128*2)
  CARVE(float*,     xsf,    1280*1024*4)
  CARVE(float*,     xsr,    1280*1024*4)
  CARVE(float*,     l0f,    1280*512*4)
  CARVE(_Float16*,  l0h,    1280*512*2)
  CARVE(float*,     l1f,    1280*512*4)
  CARVE(_Float16*,  l1h,    1280*512*2)
  CARVE(float*,     gbuf,   2*64*1024*4)
  CARVE(_Float16*,  wih0h,  1024*128*2)
  CARVE(_Float16*,  wih0rh, 1024*128*2)
  CARVE(_Float16*,  wih1h,  1024*512*2)
  CARVE(_Float16*,  wih1rh, 1024*512*2)
  CARVE(_Float16*,  whh0h,  1024*256*2)
  CARVE(_Float16*,  whh0rh, 1024*256*2)
  CARVE(_Float16*,  whh1h,  1024*256*2)
  CARVE(_Float16*,  whh1rh, 1024*256*2)
#undef CARVE
  float* out = (float*)d_out;

  // -------- pre-convert LSTM weights to f16 --------
  cvt_f16_kernel<<<(1024*128 + 255)/256, 256, 0, stream>>>(w_ih_l0,  wih0h,  1024*128);
  cvt_f16_kernel<<<(1024*128 + 255)/256, 256, 0, stream>>>(w_ih_l0r, wih0rh, 1024*128);
  cvt_f16_kernel<<<(1024*512 + 255)/256, 256, 0, stream>>>(w_ih_l1,  wih1h,  1024*512);
  cvt_f16_kernel<<<(1024*512 + 255)/256, 256, 0, stream>>>(w_ih_l1r, wih1rh, 1024*512);
  cvt_f16_kernel<<<(1024*256 + 255)/256, 256, 0, stream>>>(w_hh_l0,  whh0h,  1024*256);
  cvt_f16_kernel<<<(1024*256 + 255)/256, 256, 0, stream>>>(w_hh_l0r, whh0rh, 1024*256);
  cvt_f16_kernel<<<(1024*256 + 255)/256, 256, 0, stream>>>(w_hh_l1,  whh1h,  1024*256);
  cvt_f16_kernel<<<(1024*256 + 255)/256, 256, 0, stream>>>(w_hh_l1r, whh1rh, 1024*256);

  build_adj_kernel<<<1, 64, 0, stream>>>(edge, adj);

  conv_stack_kernel<<<10240, 256, CONV_SMEM, stream>>>(
      x, conv1_w, conv1_b, bn1_g, bn1_b,
      conv2_w, conv2_b, bn2_g, bn2_b,
      conv3_w, conv3_b, bn3_g, bn3_b, feats);

  gcn_kernel<<<1280, 128, 0, stream>>>(feats, adj, gcn0_w, gcn0_b,
                                       gcn1_w, gcn1_b, gln_g, gln_b, emb_h);

  dim3 gproj(40, 16);
  xproj_kernel<<<gproj, 256, 0, stream>>>(emb_h, wih0h,  b_ih_l0,  b_hh_l0,  xsf, 128);
  xproj_kernel<<<gproj, 256, 0, stream>>>(emb_h, wih0rh, b_ih_l0r, b_hh_l0r, xsr, 128);
  lstm_rec_kernel<<<2, 1024, LSTM_SMEM, stream>>>(xsf, xsr, whh0h, whh0rh, gbuf, l0f, l0h);

  xproj_kernel<<<gproj, 256, 0, stream>>>(l0h, wih1h,  b_ih_l1,  b_hh_l1,  xsf, 512);
  xproj_kernel<<<gproj, 256, 0, stream>>>(l0h, wih1rh, b_ih_l1r, b_hh_l1r, xsr, 512);
  lstm_rec_kernel<<<2, 1024, LSTM_SMEM, stream>>>(xsf, xsr, whh1h, whh1rh, gbuf, l1f, l1h);

  head_kernel<<<1280, 128, 0, stream>>>(l1f, lln_g, lln_b, cls_w, cls_b, out);
}
